// ProGraMLNetPyG_1717986918493
// MI455X (gfx1250) — matrix operand order repaired and checked by reading, not compiled, not run on hardware
//
#include <hip/hip_runtime.h>
#include <hip/hip_bf16.h>

typedef __bf16 bf16;
typedef __attribute__((ext_vector_type(16))) __bf16 v16bf;
typedef __attribute__((ext_vector_type(8)))  float  v8f;
typedef __attribute__((ext_vector_type(4)))  float  f32x4;
typedef __attribute__((ext_vector_type(4)))  __bf16 bf16x4;

#define N_NODES 100000
#define N_EDGES 640000

// ---------------------------------------------------------------------------
// Fast activations on the hardware TRANS pipe (co-executes with WMMA).
__device__ __forceinline__ float fast_sigmoid(float x) {
  float e = __builtin_amdgcn_exp2f(-1.442695041f * x);   // exp(-x)
  return __builtin_amdgcn_rcpf(1.f + e);                 // 1/(1+exp(-x))
}
__device__ __forceinline__ float fast_tanh(float x) {
  float a = fabsf(x);
  float e = __builtin_amdgcn_exp2f(-2.885390082f * a);   // exp(-2|x|), in (0,1]
  float r = (1.f - e) * __builtin_amdgcn_rcpf(1.f + e);
  return copysignf(r, x);
}
__device__ __forceinline__ bf16x4 cvt4(f32x4 v) {
  bf16x4 r;
  r[0] = (bf16)v[0]; r[1] = (bf16)v[1]; r[2] = (bf16)v[2]; r[3] = (bf16)v[3];
  return r;
}

// ---------------------------------------------------------------------------
// WMMA fragment loaders (CDNA5 16x16x32 bf16 layouts, wave32)
// A tile: [16 rows][stride] bf16 in LDS, row-major -> 2x ds_load_b128 per lane.
__device__ __forceinline__ v16bf load_A(const bf16* sA, int stride, int kc, int lane) {
  int m  = lane & 15;
  int kb = kc + ((lane & 16) ? 16 : 0);
  const bf16* p = sA + m * stride + kb;
  v16bf a;
#pragma unroll
  for (int j = 0; j < 8; ++j) a[j] = p[j];
#pragma unroll
  for (int j = 0; j < 8; ++j) a[8 + j] = p[8 + j];
  return a;
}

// B fragment straight from a global row-major [K][ldn] f32 matrix (one-time /
// loop-invariant loads that live in registers; L2-resident weights).
__device__ __forceinline__ v16bf load_B_g(const float* __restrict__ W, int ldn,
                                          int kc, int nb, int lane) {
  int col = nb + (lane & 15);
  int kb  = kc + (lane & 16);
  const float* p = W + (size_t)kb * ldn + col;
  v16bf b;
#pragma unroll
  for (int j = 0; j < 16; ++j) b[j] = (bf16)p[(size_t)j * ldn];
  return b;
}

// B fragment from LDS pre-swizzled into fragment order (32 contiguous B/lane).
__device__ __forceinline__ v16bf load_B_sw(const bf16* sB, int ncols, int kc, int nb, int lane) {
  int col = nb + (lane & 15);
  const bf16* p = sB + ((size_t)((kc >> 5) * ncols + col) << 5) + ((lane & 16) ? 16 : 0);
  v16bf b;
#pragma unroll
  for (int j = 0; j < 16; ++j) b[j] = p[j];
  return b;
}

__device__ __forceinline__ v8f wmma_bf16(v16bf a, v16bf b, v8f c) {
  return __builtin_amdgcn_wmma_f32_16x16x32_bf16(false, a, false, b, (short)0, c, false, false);
}

// ---------------------------------------------------------------------------
// Utility kernels
__global__ void fill_f32(float* p, long n, float v) {
  long i = (long)blockIdx.x * blockDim.x + threadIdx.x;
  if (i < n) p[i] = v;
}
__global__ void fill_f32x4(f32x4* p, long n4) {
  long i = (long)blockIdx.x * blockDim.x + threadIdx.x;
  if (i < n4) p[i] = (f32x4){};
}
__global__ void fill_i32(int* p, long n, int v) {
  long i = (long)blockIdx.x * blockDim.x + threadIdx.x;
  if (i < n) p[i] = v;
}

__global__ void edge_prep(const int* __restrict__ ei, const int* __restrict__ et, int E,
                          int* __restrict__ tcount, float* __restrict__ cnt) {
  int e = blockIdx.x * blockDim.x + threadIdx.x;
  if (e < E) {
    atomicAdd(&tcount[et[e]], 1);
    atomicAdd(&cnt[ei[E + e]], 1.0f);
  }
}

// meta[0..3]: cumulative 16-edge tile counts per type; meta[4..6]: perm start per type
__global__ void calc_offsets(const int* __restrict__ tcount, int* __restrict__ meta) {
  int tb = 0, ps = 0;
  meta[0] = 0;
#pragma unroll
  for (int t = 0; t < 3; ++t) {
    int c = tcount[t];
    int tiles = (c + 15) >> 4;
    meta[4 + t] = ps;
    ps += tiles * 16;
    tb += tiles;
    meta[1 + t] = tb;
  }
}

__global__ void scatter_perm(const int* __restrict__ et, int E, const int* __restrict__ meta,
                             int* __restrict__ fillc, int* __restrict__ perm) {
  int e = blockIdx.x * blockDim.x + threadIdx.x;
  if (e < E) {
    int t = et[e];
    int pos = meta[4 + t] + atomicAdd(&fillc[t], 1);
    perm[pos] = e;
  }
}

// h0 = [emb[idx] | selectors], built 16B at a time (rows are 16B aligned)
__global__ void build_h0(const int* __restrict__ idx, const f32x4* __restrict__ sel,
                         const f32x4* __restrict__ emb, f32x4* __restrict__ h0, int N) {
  long i = (long)blockIdx.x * blockDim.x + threadIdx.x;
  if (i < (long)N * 32) {
    int n = (int)(i >> 5), q = (int)(i & 31);
    f32x4 v = (q < 24) ? emb[(size_t)idx[n] * 24 + q] : sel[(size_t)n * 8 + (q - 24)];
    h0[i] = v;
  }
}

// ---------------------------------------------------------------------------
// Edge message kernel. Weight B-fragments in registers (loop-invariant per
// wave); ~14.5 KB LDS so many blocks co-reside per WGP to hide gather/scatter
// latency. All global staging moves 16 B per thread (global_load_b128).
__global__ void __launch_bounds__(256) msg_kernel(
    const float* __restrict__ h, const int* __restrict__ ei, const int* __restrict__ epos,
    const float* __restrict__ pos_enc, const float* __restrict__ Wg, const float* __restrict__ bg,
    const float* __restrict__ Wt, const float* __restrict__ bt,
    const int* __restrict__ perm, const int* __restrict__ meta,
    float* __restrict__ agg, int E) {
  __shared__ __align__(16) bf16  sPE[16 * 64];     // 2 KB
  __shared__ __align__(16) float sGate[16 * 128];  // 8 KB
  __shared__ __align__(16) bf16  sA[16 * 128];     // 4 KB
  __shared__ int sSrc[16], sDst[16], sPos[16];

  const int tid = threadIdx.x, lane = tid & 31, wid = tid >> 5;
  const int nb = wid * 16;
  const int nn = nb + (lane & 15);
  const int mofs = (lane & 16) ? 8 : 0;

  v16bf wgf[2];
#pragma unroll
  for (int c = 0; c < 2; ++c) wgf[c] = load_B_g(Wg, 128, c * 32, nb, lane);
  const float bgv = bg[nn];

  const int tb1 = meta[1], tb2 = meta[2], tb3 = meta[3];
  int cur = -1;
  v16bf wtf[4];
  float btv = 0.f;

  for (int tile = blockIdx.x; tile < tb3; tile += gridDim.x) {
    int t = (tile < tb1) ? 0 : ((tile < tb2) ? 1 : 2);
    if (t != cur) {  // uniform; registers only, no barrier needed
      const float* w = Wt + (size_t)t * 128 * 128;
#pragma unroll
      for (int c = 0; c < 4; ++c) wtf[c] = load_B_g(w, 128, c * 32, nb, lane);
      btv = bt[t * 128 + nn];
      cur = t;
    }
    int eb = meta[4 + t] + (tile - meta[t]) * 16;  // meta[0]==0
    if (tid < 16) {
      int e = perm[eb + tid];
      sSrc[tid] = (e < 0) ? -1 : ei[e];
      sDst[tid] = (e < 0) ? -1 : ei[E + e];
      sPos[tid] = (e < 0) ? -1 : epos[e];
    }
    __syncthreads();
    {  // pe tile: 256 f32x4 slots, exactly one per thread
      int m = tid >> 4, q = tid & 15;
      int p = sPos[m];
      f32x4 v = {};
      if (p >= 0) v = ((const f32x4*)pos_enc)[(size_t)p * 16 + q];
      ((bf16x4*)sPE)[tid] = cvt4(v);
    }
    __syncthreads();
    // gate = 2*sigmoid(pe @ Wg + bg): 2 WMMAs per wave (K=64)
    {
      v8f acc = {};
#pragma unroll
      for (int c = 0; c < 2; ++c)
        acc = wmma_bf16(load_A(sPE, 64, c * 32, lane), wgf[c], acc);
#pragma unroll
      for (int r = 0; r < 8; ++r)
        sGate[(r + mofs) * 128 + nn] = 2.f * fast_sigmoid(acc[r] + bgv);
    }
    __syncthreads();
    // A = bf16( h[src] * gate ), 16 B loads per thread
#pragma unroll
    for (int i = tid; i < 16 * 32; i += 256) {
      int m = i >> 5;
      int s = sSrc[m];
      f32x4 hv = {};
      if (s >= 0) hv = ((const f32x4*)h)[(size_t)s * 32 + (i & 31)];
      f32x4 g = ((const f32x4*)sGate)[i];
      ((bf16x4*)sA)[i] = cvt4(hv * g);
    }
    __syncthreads();
    // msg = A @ Wt[t] + bt[t], scattered to agg[dst]
    {
      v8f acc = {};
#pragma unroll
      for (int c = 0; c < 4; ++c)
        acc = wmma_bf16(load_A(sA, 128, c * 32, lane), wtf[c], acc);
#pragma unroll
      for (int r = 0; r < 8; ++r) {
        int d = sDst[r + mofs];
        if (d >= 0) atomicAdd(&agg[(size_t)d * 128 + nn], acc[r] + btv);
      }
    }
    __syncthreads();
  }
}

// ---------------------------------------------------------------------------
// Fused GRU cell. W_ih^T / W_hh^T resident in LDS, pre-swizzled to fragment
// order so each fragment read is 2x ds_load_b128. ~229 KB dynamic LDS.
__global__ void __launch_bounds__(256) gru_kernel(
    float* __restrict__ h, const float* __restrict__ agg, const float* __restrict__ cnt,
    const float* __restrict__ Wih, const float* __restrict__ Whh,
    const float* __restrict__ bih, const float* __restrict__ bhh, int N) {
  extern __shared__ __align__(16) char smem[];
  bf16*  sBih = (bf16*)smem;             // swizzled [4][384][32]
  bf16*  sBhh = sBih + 128 * 384;
  bf16*  sA1  = sBhh + 128 * 384;        // [16][128] agg/den
  bf16*  sA2  = sA1 + 16 * 128;          // [16][128] h
  float* sH   = (float*)(sA2 + 16 * 128);
  float* sR   = sH + 16 * 128;
  float* sZ   = sR + 16 * 128;

  const int tid = threadIdx.x, lane = tid & 31, wid = tid >> 5;
  const int c0 = wid * 16 + (lane & 15);
  const int mofs = (lane & 16) ? 8 : 0;

  float biR[3], bhR[3];
#pragma unroll
  for (int ph = 0; ph < 3; ++ph) { biR[ph] = bih[c0 + ph * 128]; bhR[ph] = bhh[c0 + ph * 128]; }

  // swizzle-load weights: sw[((c*384 + n)<<5) + h16*16 + j] = W^T[c*32+h16*16+j][n]
  for (int i = tid; i < 128 * 384; i += 256) {
    int j = i & 15, h16 = (i >> 4) & 1;
    int n = (i >> 5) % 384, c = (i >> 5) / 384;
    int k = c * 32 + h16 * 16 + j;
    size_t g = (size_t)n * 128 + k;          // B[k][n] = W[n][k]
    sBih[i] = (bf16)Wih[g];
    sBhh[i] = (bf16)Whh[g];
  }

  int ntiles = (N + 15) >> 4;
  for (int tile = blockIdx.x; tile < ntiles; tile += gridDim.x) {
    __syncthreads();
    int n0 = tile * 16;
#pragma unroll
    for (int i = tid; i < 16 * 32; i += 256) {   // 16 B loads per thread
      int m = i >> 5, q = i & 31;
      int node = n0 + m;
      f32x4 hv = {}, av = {};
      if (node < N) {
        hv = ((const f32x4*)h)[(size_t)node * 32 + q];
        float c = cnt[node];
        float rd = __builtin_amdgcn_rcpf(c > 1.f ? c : 1.f);
        av = ((const f32x4*)agg)[(size_t)node * 32 + q] * rd;
      }
      ((f32x4*)sH)[i] = hv;
      ((bf16x4*)sA2)[i] = cvt4(hv);
      ((bf16x4*)sA1)[i] = cvt4(av);
    }
    __syncthreads();
    // phase 1: r (cols 0-127, j=wid) and z (cols 128-255, j=wid+8)
#pragma unroll
    for (int ph = 0; ph < 2; ++ph) {
      int nb = (wid + ph * 8) * 16;
      v8f ai = {}, ah = {};
#pragma unroll
      for (int kc = 0; kc < 128; kc += 32) {
        ai = wmma_bf16(load_A(sA1, 128, kc, lane), load_B_sw(sBih, 384, kc, nb, lane), ai);
        ah = wmma_bf16(load_A(sA2, 128, kc, lane), load_B_sw(sBhh, 384, kc, nb, lane), ah);
      }
      float* dst = ph ? sZ : sR;
#pragma unroll
      for (int r = 0; r < 8; ++r)
        dst[(r + mofs) * 128 + c0] = fast_sigmoid(ai[r] + biR[ph] + ah[r] + bhR[ph]);
    }
    __syncthreads();
    // phase 2: n gate (cols 256-383, j=wid+16) + blend + in-place store
    {
      int nb = (wid + 16) * 16;
      v8f ai = {}, ah = {};
#pragma unroll
      for (int kc = 0; kc < 128; kc += 32) {
        ai = wmma_bf16(load_A(sA1, 128, kc, lane), load_B_sw(sBih, 384, kc, nb, lane), ai);
        ah = wmma_bf16(load_A(sA2, 128, kc, lane), load_B_sw(sBhh, 384, kc, nb, lane), ah);
      }
#pragma unroll
      for (int r = 0; r < 8; ++r) {
        int m = r + mofs;
        int node = n0 + m;
        float i_n = ai[r] + biR[2];
        float h_n = ah[r] + bhR[2];
        float nv = fast_tanh(i_n + sR[m * 128 + c0] * h_n);
        float zz = sZ[m * 128 + c0];
        float hn = (1.f - zz) * nv + zz * sH[m * 128 + c0];
        if (node < N) h[(size_t)node * 128 + c0] = hn;
      }
    }
  }
}

// ---------------------------------------------------------------------------
// Head: logits = relu([h,h0] @ W1 + b1) @ W2 + b2. W1 fragments in registers.
__global__ void __launch_bounds__(256) head_kernel(
    const float* __restrict__ h, const float* __restrict__ h0,
    const float* __restrict__ W1, const float* __restrict__ b1,
    const float* __restrict__ W2, const float* __restrict__ b2,
    float* __restrict__ out, int N) {
  __shared__ __align__(16) bf16  sA[16 * 256];     // 8 KB
  __shared__ __align__(16) float sHid[16 * 128];   // 8 KB
  __shared__ float sW2[128];

  const int tid = threadIdx.x, lane = tid & 31, wid = tid >> 5;
  const int nb = wid * 16;
  const int nn = nb + (lane & 15);
  const int mofs = (lane & 16) ? 8 : 0;

  v16bf b1f[8];
#pragma unroll
  for (int c = 0; c < 8; ++c) b1f[c] = load_B_g(W1, 128, c * 32, nb, lane);
  const float b1v = b1[nn];
  const float b2v = b2[0];
  for (int i = tid; i < 128; i += 256) sW2[i] = W2[i];

  int ntiles = (N + 15) >> 4;
  for (int tile = blockIdx.x; tile < ntiles; tile += gridDim.x) {
    __syncthreads();
    int n0 = tile * 16;
#pragma unroll
    for (int i = tid; i < 16 * 64; i += 256) {   // feats: 16 B loads per thread
      int m = i >> 6, q = i & 63;
      int node = n0 + m;
      f32x4 v = {};
      if (node < N)
        v = (q < 32) ? ((const f32x4*)h)[(size_t)node * 32 + q]
                     : ((const f32x4*)h0)[(size_t)node * 32 + (q - 32)];
      ((bf16x4*)sA)[i] = cvt4(v);
    }
    __syncthreads();
    {
      v8f acc = {};
#pragma unroll
      for (int c = 0; c < 8; ++c)
        acc = wmma_bf16(load_A(sA, 256, c * 32, lane), b1f[c], acc);
#pragma unroll
      for (int r = 0; r < 8; ++r) {
        float x = acc[r] + b1v;
        sHid[(r + mofs) * 128 + nn] = x > 0.f ? x : 0.f;
      }
    }
    __syncthreads();
    if (tid < 16) {
      int node = n0 + tid;
      if (node < N) {
        float s = 0.f;
#pragma unroll 8
        for (int k = 0; k < 128; ++k) s += sHid[tid * 128 + k] * sW2[k];
        out[node] = s + b2v;
      }
    }
  }
}

// ---------------------------------------------------------------------------
extern "C" void kernel_launch(void* const* d_in, const int* in_sizes, int n_in,
                              void* d_out, int out_size, void* d_ws, size_t ws_size,
                              hipStream_t stream) {
  const int N = N_NODES, E = N_EDGES;
  const int*   x_idx = (const int*)d_in[0];
  const float* sel   = (const float*)d_in[1];
  const int*   ei    = (const int*)d_in[2];
  const int*   et    = (const int*)d_in[3];
  const int*   ep    = (const int*)d_in[4];
  const float* emb   = (const float*)d_in[5];
  const float* pe    = (const float*)d_in[6];
  const float* Wg    = (const float*)d_in[7];
  const float* bg    = (const float*)d_in[8];
  const float* Wt    = (const float*)d_in[9];
  const float* bt    = (const float*)d_in[10];
  const float* Wih   = (const float*)d_in[11];
  const float* Whh   = (const float*)d_in[12];
  const float* bih   = (const float*)d_in[13];
  const float* bhh   = (const float*)d_in[14];
  const float* W1    = (const float*)d_in[15];
  const float* b1    = (const float*)d_in[16];
  const float* W2    = (const float*)d_in[17];
  const float* b2    = (const float*)d_in[18];
  float* out = (float*)d_out;

  char* ws = (char*)d_ws;
  size_t off = 0;
  auto alloc = [&](size_t bytes) {
    void* p = ws + off;
    off = (off + bytes + 255) & ~(size_t)255;
    return p;
  };
  float* h0   = (float*)alloc((size_t)N * 128 * 4);
  float* h    = (float*)alloc((size_t)N * 128 * 4);
  float* agg  = (float*)alloc((size_t)N * 128 * 4);
  float* cnt  = (float*)alloc((size_t)N * 4);
  int*   perm = (int*)alloc((size_t)(E + 64) * 4);
  int*   meta = (int*)alloc(64 * 4);
  int* tcount = meta + 8;
  int* fillc  = meta + 12;

  const size_t gruSmem = (size_t)(2 * 128 * 384 + 2 * 16 * 128) * 2 +
                         (size_t)(3 * 16 * 128) * 4;  // ~229 KB
  (void)hipFuncSetAttribute((const void*)gru_kernel,
                            hipFuncAttributeMaxDynamicSharedMemorySize, (int)gruSmem);

  // ---- preprocessing (deterministic per-call) ----
  fill_f32<<<(N + 255) / 256, 256, 0, stream>>>(cnt, N, 0.f);
  fill_i32<<<(E + 64 + 255) / 256, 256, 0, stream>>>(perm, E + 64, -1);
  fill_i32<<<1, 64, 0, stream>>>(meta, 16, 0);
  edge_prep<<<(E + 255) / 256, 256, 0, stream>>>(ei, et, E, tcount, cnt);
  calc_offsets<<<1, 1, 0, stream>>>(tcount, meta);
  scatter_perm<<<(E + 255) / 256, 256, 0, stream>>>(et, E, meta, fillc, perm);
  build_h0<<<(int)(((size_t)N * 32 + 255) / 256), 256, 0, stream>>>(
      x_idx, (const f32x4*)sel, (const f32x4*)emb, (f32x4*)h0, N);
  hipMemcpyAsync(h, h0, (size_t)N * 128 * 4, hipMemcpyDeviceToDevice, stream);

  // ---- 6 GGNN iterations ----
  for (int it = 0; it < 6; ++it) {
    fill_f32x4<<<(int)(((size_t)N * 32 + 255) / 256), 256, 0, stream>>>(
        (f32x4*)agg, (long)N * 32);
    msg_kernel<<<4096, 256, 0, stream>>>(h, ei, ep, pe, Wg, bg, Wt, bt, perm, meta, agg, E);
    gru_kernel<<<2048, 256, gruSmem, stream>>>(h, agg, cnt, Wih, Whh, bih, bhh, N);
  }

  // ---- head ----
  head_kernel<<<2048, 256, 0, stream>>>(h, h0, W1, b1, W2, b2, out, N);
}
